// TSPGNN_16724602650929
// MI455X (gfx1250) — compile-verified, matrix operand order checked by reading, not run on hardware
//
#include <hip/hip_runtime.h>
#include <math.h>

// ---------------- constants ----------------
#define Hd 300
#define NKT 10            // K tiles of 32 (K padded to 320)
#define NNT 19            // out-col tiles of 16 (out padded to 304)
#define Nn 16000
#define Ne 256000
#define Lg 3
#define ROWT (Nn / 16)    // 1000 row tiles
#define MT 4              // row tiles per wave in GEMM (64 rows)
#define ETILES (Ne / 16)  // 16000 edge tiles
#define WSLOT (NNT * NKT * 32 * 16)       // u16 elems per packed weight (97280)
#define ATSTRIDE ((size_t)NKT * 32 * 16)  // u16 elems per A row-tile (5120)
#define KSTEP (32 * 16)                   // u16 elems per k-tile fragment (512)

typedef __attribute__((ext_vector_type(16))) __bf16 v16bf;
typedef __attribute__((ext_vector_type(8)))  float  v8f;
typedef unsigned short u16;
typedef unsigned int   u32;

// 16-bit A/B fragment K index for element e (0..15) given lane half (ISA 7.12.2)
__device__ __forceinline__ int klocal(int e, int half) {
  int v = e >> 1;
  int base = (v < 4) ? (2 * v) : (16 + 2 * (v - 4));
  return base + 8 * half + (e & 1);
}

__device__ __forceinline__ u16 f2bf(float f) {
  u32 u = __float_as_uint(f);
  return (u16)((u + 0x7FFFu + ((u >> 16) & 1u)) >> 16);  // RNE
}

// order-preserving float<->uint for atomic max
__device__ __forceinline__ u32 fenc(float f) {
  u32 u = __float_as_uint(f);
  return (u & 0x80000000u) ? ~u : (u | 0x80000000u);
}
__device__ __forceinline__ float fdec(u32 u) {
  u32 v = (u & 0x80000000u) ? (u & 0x7FFFFFFFu) : ~u;
  return __uint_as_float(v);
}

// ---------------- utility kernels ----------------
__global__ void fill_f32(float* p, float v, int n) {
  int i = blockIdx.x * blockDim.x + threadIdx.x;
  if (i < n) p[i] = v;
}
__global__ void fill_u32(u32* p, u32 v, int n) {
  int i = blockIdx.x * blockDim.x + threadIdx.x;
  if (i < n) p[i] = v;
}
__global__ void relu_k(float* p, int n) {
  int i = blockIdx.x * blockDim.x + threadIdx.x;
  if (i < n) { float v = p[i]; p[i] = v > 0.f ? v : 0.f; }
}

// h = x @ W_in + b_in   (x is [N,2])
__global__ void init_embed(const float* x, const float* Win, const float* bin, float* h) {
  int i = blockIdx.x * blockDim.x + threadIdx.x;
  if (i >= Nn * Hd) return;
  int n = i / Hd, hh = i - n * Hd;
  h[i] = x[n * 2] * Win[hh] + x[n * 2 + 1] * Win[Hd + hh] + bin[hh];
}

// rank-1 collapse of e@gWE: u[l][n] = sum_k W_edge[k]*gWE[l,k,n]; c[l][n] = sum_k b_edge[k]*gWE[l,k,n] + gbE[l,n]
__global__ void edge_uc(const float* Wedge, const float* bedge, const float* gWE,
                        const float* gbE, float* u, float* c) {
  int i = blockIdx.x * blockDim.x + threadIdx.x;
  if (i >= Lg * Hd) return;
  int l = i / Hd, n = i - l * Hd;
  const float* W = gWE + (size_t)l * Hd * Hd;
  float su = 0.f, sc = 0.f;
  for (int k = 0; k < Hd; ++k) { float w = W[k * Hd + n]; su += Wedge[k] * w; sc += bedge[k] * w; }
  u[i] = su; c[i] = sc + gbE[i];
}

// pack weight [300,300] row-major (k,n) into B fragments: dst[((nt*NKT+kt)*32+lane)*16+e]
__global__ void pack_w(const float* W, u16* dst) {
  int i = blockIdx.x * blockDim.x + threadIdx.x;
  if (i >= NNT * NKT * 32) return;
  int lane = i & 31, kt = (i >> 5) % NKT, nt = i / (32 * NKT);
  int half = lane >> 4, n = nt * 16 + (lane & 15);
  u16* d = dst + (size_t)i * 16;
#pragma unroll
  for (int e = 0; e < 16; ++e) {
    int k = kt * 32 + klocal(e, half);
    float v = (k < Hd && n < Hd) ? W[k * Hd + n] : 0.f;
    d[e] = f2bf(v);
  }
}

// pack h [16000,300] into A fragments: dst[((rt*NKT+kt)*32+lane)*16+e]
__global__ void pack_a(const float* h, u16* dst) {
  int i = blockIdx.x * blockDim.x + threadIdx.x;
  if (i >= ROWT * NKT * 32) return;
  int lane = i & 31, kt = (i >> 5) % NKT, rt = i / (32 * NKT);
  int half = lane >> 4, m = rt * 16 + (lane & 15);
  const float* hr = h + (size_t)m * Hd;
  u16* d = dst + (size_t)i * 16;
#pragma unroll
  for (int e = 0; e < 16; ++e) {
    int k = kt * 32 + klocal(e, half);
    d[e] = f2bf(k < Hd ? hr[k] : 0.f);
  }
}

// out[M,300] = A(bf16 packed) @ B(bf16 packed) + bias
// one wave per 64x16 output block; explicit double-buffering of the next
// k-tile's B + 4xA fragments so WMMAs overlap the loads.
__global__ void gemm16(const u16* __restrict__ Ap, const u16* __restrict__ Bp,
                       const float* __restrict__ bias, float* __restrict__ out) {
  int wave = threadIdx.x >> 5, lane = threadIdx.x & 31;
  int rb = blockIdx.x;               // 250 row blocks of 64 rows
  int nt = blockIdx.y * 8 + wave;
  if (nt >= NNT) return;             // wave-uniform
  const u16* bptr = Bp + ((size_t)(nt * NKT) * 32 + lane) * 16;
  const u16* aptr = Ap + ((size_t)(rb * MT * NKT) * 32 + lane) * 16;

  v16bf b  = *reinterpret_cast<const v16bf*>(bptr);
  v16bf x0 = *reinterpret_cast<const v16bf*>(aptr);
  v16bf x1 = *reinterpret_cast<const v16bf*>(aptr + ATSTRIDE);
  v16bf x2 = *reinterpret_cast<const v16bf*>(aptr + 2 * ATSTRIDE);
  v16bf x3 = *reinterpret_cast<const v16bf*>(aptr + 3 * ATSTRIDE);

  v8f c0 = {}, c1 = {}, c2 = {}, c3 = {};
#pragma unroll
  for (int kt = 0; kt < NKT; ++kt) {
    v16bf bn = b, y0 = x0, y1 = x1, y2 = x2, y3 = x3;
    if (kt + 1 < NKT) {
      const u16* bp2 = bptr + (size_t)(kt + 1) * KSTEP;
      const u16* ap2 = aptr + (size_t)(kt + 1) * KSTEP;
      bn = *reinterpret_cast<const v16bf*>(bp2);
      y0 = *reinterpret_cast<const v16bf*>(ap2);
      y1 = *reinterpret_cast<const v16bf*>(ap2 + ATSTRIDE);
      y2 = *reinterpret_cast<const v16bf*>(ap2 + 2 * ATSTRIDE);
      y3 = *reinterpret_cast<const v16bf*>(ap2 + 3 * ATSTRIDE);
    }
    c0 = __builtin_amdgcn_wmma_f32_16x16x32_bf16(false, x0, false, b, (short)0, c0, false, false);
    c1 = __builtin_amdgcn_wmma_f32_16x16x32_bf16(false, x1, false, b, (short)0, c1, false, false);
    c2 = __builtin_amdgcn_wmma_f32_16x16x32_bf16(false, x2, false, b, (short)0, c2, false, false);
    c3 = __builtin_amdgcn_wmma_f32_16x16x32_bf16(false, x3, false, b, (short)0, c3, false, false);
    b = bn; x0 = y0; x1 = y1; x2 = y2; x3 = y3;
  }

  int half = lane >> 4, n = nt * 16 + (lane & 15);
  if (n < Hd) {
    float bb = bias[n];
    int m0 = rb * MT * 16 + half * 8;
#pragma unroll
    for (int i = 0; i < 8; ++i) {
      out[(size_t)(m0 + i) * Hd + n]      = c0[i] + bb;
      out[(size_t)(m0 + 16 + i) * Hd + n] = c1[i] + bb;
      out[(size_t)(m0 + 32 + i) * Hd + n] = c2[i] + bb;
      out[(size_t)(m0 + 48 + i) * Hd + n] = c3[i] + bb;
    }
  }
}

// logits[e] = dot(q[col], k[row]) / sqrt(H) ; one wave per edge
__global__ void attn_logits(const float* q, const float* kk, const int* ei, float* lg) {
  int w = (blockIdx.x * blockDim.x + threadIdx.x) >> 5;
  int lane = threadIdx.x & 31;
  if (w >= Ne) return;
  int r = ei[w], c = ei[Ne + w];
  const float* qv = q + (size_t)c * Hd;
  const float* kv = kk + (size_t)r * Hd;
  float s = 0.f;
  for (int i = lane; i < Hd; i += 32) s += qv[i] * kv[i];
#pragma unroll
  for (int off = 16; off; off >>= 1) s += __shfl_xor(s, off, 32);
  if (lane == 0) lg[w] = s * 0.05773502691896258f;
}

__global__ void seg_max(const float* lg, const int* ei, u32* mE) {
  int e = blockIdx.x * blockDim.x + threadIdx.x;
  if (e >= Ne) return;
  atomicMax(&mE[ei[Ne + e]], fenc(lg[e]));
}
__global__ void seg_expsum(float* lg, const int* ei, const u32* mE, float* s) {
  int e = blockIdx.x * blockDim.x + threadIdx.x;
  if (e >= Ne) return;
  int c = ei[Ne + e];
  float ex = expf(lg[e] - fdec(mE[c]));
  lg[e] = ex;
  atomicAdd(&s[c], ex);
}
__global__ void alpha_div(float* lg, const int* ei, const float* s) {
  int e = blockIdx.x * blockDim.x + threadIdx.x;
  if (e >= Ne) return;
  lg[e] = lg[e] / s[ei[Ne + e]];
}
__global__ void attn_agg(const float* alpha, const float* v, const int* ei, float* hn) {
  int i = blockIdx.x * blockDim.x + threadIdx.x;
  if (i >= Ne * Hd) return;
  int e = i / Hd, hh = i - e * Hd;
  atomicAdd(&hn[(size_t)ei[Ne + e] * Hd + hh], alpha[e] * v[(size_t)ei[e] * Hd + hh]);
}

// gated edge: sigmoid(Bn[row]+Cn[col]+ea*u+c) * An[row] scatter-added to hn[col]
__global__ void gated_edge(const float* An, const float* Bn, const float* Cn,
                           const float* ea, const float* u, const float* cv,
                           const int* ei, float* hn) {
  int i = blockIdx.x * blockDim.x + threadIdx.x;
  if (i >= Ne * Hd) return;
  int e = i / Hd, hh = i - e * Hd;
  int r = ei[e], c = ei[Ne + e];
  float t = Bn[(size_t)r * Hd + hh] + Cn[(size_t)c * Hd + hh] + ea[e] * u[hh] + cv[hh];
  float g = 1.f / (1.f + expf(-t));
  atomicAdd(&hn[(size_t)c * Hd + hh], g * An[(size_t)r * Hd + hh]);
}

// fused edge MLP: scores = relu(|h[row]-h[col]| @ mW1 + mb1) @ mW2 + mb2, 16 edges/block
__global__ void edge_mlp(const float* __restrict__ h, const int* __restrict__ ei,
                         const u16* __restrict__ W1p, const float* __restrict__ mb1,
                         const float* __restrict__ mW2, const float* __restrict__ mb2,
                         float* __restrict__ out) {
  __shared__ __align__(32) u16 aLds[NKT * 32 * 16];
  __shared__ float rowsum[16];
  __shared__ int rS[16], cS[16];
  int tile = blockIdx.x, t = threadIdx.x;
  if (t < 16) {
    rowsum[t] = 0.f;
    rS[t] = ei[tile * 16 + t];
    cS[t] = ei[Ne + tile * 16 + t];
  }
  __syncthreads();
  // build |h[row]-h[col]| directly in A-fragment layout
  for (int slot = t; slot < NKT * 32; slot += 256) {
    int kt = slot >> 5, lane = slot & 31;
    int m = lane & 15, half = lane >> 4;
    const float* hr = h + (size_t)rS[m] * Hd;
    const float* hc = h + (size_t)cS[m] * Hd;
    u16* d = aLds + slot * 16;
#pragma unroll
    for (int e = 0; e < 16; ++e) {
      int k = kt * 32 + klocal(e, half);
      d[e] = f2bf(k < Hd ? fabsf(hr[k] - hc[k]) : 0.f);
    }
  }
  __syncthreads();
  int wave = t >> 5, lane = t & 31;
  int half = lane >> 4, nl = lane & 15;
  for (int nt = wave; nt < NNT; nt += 8) {  // wave-uniform loop
    v8f c = {};
    for (int kt = 0; kt < NKT; ++kt) {
      v16bf a = *reinterpret_cast<const v16bf*>(aLds + (kt * 32 + lane) * 16);
      v16bf b = *reinterpret_cast<const v16bf*>(W1p + ((size_t)(nt * NKT + kt) * 32 + lane) * 16);
      c = __builtin_amdgcn_wmma_f32_16x16x32_bf16(false, a, false, b, (short)0, c, false, false);
    }
    int n = nt * 16 + nl;
    if (n < Hd) {
      float b1 = mb1[n], w2 = mW2[n];
#pragma unroll
      for (int i = 0; i < 8; ++i) {
        float r = c[i] + b1;
        r = r > 0.f ? r : 0.f;
        atomicAdd(&rowsum[half * 8 + i], r * w2);
      }
    }
  }
  __syncthreads();
  if (t < 16) out[tile * 16 + t] = rowsum[t] + mb2[0];
}

// ---------------- launch ----------------
extern "C" void kernel_launch(void* const* d_in, const int* in_sizes, int n_in,
                              void* d_out, int out_size, void* d_ws, size_t ws_size,
                              hipStream_t stream) {
  (void)in_sizes; (void)n_in; (void)out_size; (void)ws_size;
  const float* x      = (const float*)d_in[0];
  const int*   ei     = (const int*)d_in[1];
  const float* eattr  = (const float*)d_in[2];
  const float* W_in   = (const float*)d_in[3];
  const float* b_in   = (const float*)d_in[4];
  const float* W_edge = (const float*)d_in[5];
  const float* b_edge = (const float*)d_in[6];
  const float* Wq  = (const float*)d_in[7];  const float* bq  = (const float*)d_in[8];
  const float* Wk  = (const float*)d_in[9];  const float* bk  = (const float*)d_in[10];
  const float* Wv  = (const float*)d_in[11]; const float* bv  = (const float*)d_in[12];
  const float* Wsk = (const float*)d_in[13]; const float* bsk = (const float*)d_in[14];
  const float* gWA = (const float*)d_in[15]; const float* gbA = (const float*)d_in[16];
  const float* gWB = (const float*)d_in[17]; const float* gbB = (const float*)d_in[18];
  const float* gWC = (const float*)d_in[19]; const float* gbC = (const float*)d_in[20];
  const float* gWE = (const float*)d_in[21]; const float* gbE = (const float*)d_in[22];
  const float* gWr = (const float*)d_in[23]; const float* gbr = (const float*)d_in[24];
  const float* mW1 = (const float*)d_in[25]; const float* mb1 = (const float*)d_in[26];
  const float* mW2 = (const float*)d_in[27]; const float* mb2 = (const float*)d_in[28];

  char* ws = (char*)d_ws;
  float* hA   = (float*)(ws + 0);           // 19,200,000 B
  float* hB   = (float*)(ws + 19200000);
  float* Qb   = (float*)(ws + 38400000);
  float* Kb   = (float*)(ws + 57600000);
  float* Vb   = (float*)(ws + 76800000);
  u16*   Ap   = (u16*)  (ws + 96000000);    // 10,240,000 B
  u16*   Wp   = (u16*)  (ws + 106240000);   // 17 * 194,560 B
  float* lg   = (float*)(ws + 109547520);   // 1,024,000 B
  u32*   mE   = (u32*)  (ws + 110571520);   // 64,000 B
  float* sS   = (float*)(ws + 110635520);   // 64,000 B
  float* uB   = (float*)(ws + 110699520);   // 3,840 B
  float* cB   = (float*)(ws + 110703360);   // 3,840 B

  const int HH = Hd * Hd;
  const float* wsrc[17] = {Wq, Wk, Wv, Wsk,
                           gWA, gWA + HH, gWA + 2 * HH,
                           gWB, gWB + HH, gWB + 2 * HH,
                           gWC, gWC + HH, gWC + 2 * HH,
                           gWr, gWr + HH, gWr + 2 * HH,
                           mW1};
  for (int i = 0; i < 17; ++i)
    pack_w<<<(NNT * NKT * 32 + 255) / 256, 256, 0, stream>>>(wsrc[i], Wp + (size_t)i * WSLOT);

  edge_uc<<<(Lg * Hd + 255) / 256, 256, 0, stream>>>(W_edge, b_edge, gWE, gbE, uB, cB);
  init_embed<<<(Nn * Hd + 255) / 256, 256, 0, stream>>>(x, W_in, b_in, hA);

  dim3 gg(ROWT / MT, 3);                    // 250 x 3 blocks, 8 waves each
  int apGrid = (ROWT * NKT * 32 + 255) / 256;
  int eGrid  = (Ne + 255) / 256;
  int ehGrid = (Ne * Hd + 255) / 256;
  int nhGrid = (Nn * Hd + 255) / 256;

  // ---- TransformerConv ----
  pack_a<<<apGrid, 256, 0, stream>>>(hA, Ap);
  gemm16<<<gg, 256, 0, stream>>>(Ap, Wp + (size_t)0 * WSLOT, bq, Qb);
  gemm16<<<gg, 256, 0, stream>>>(Ap, Wp + (size_t)1 * WSLOT, bk, Kb);
  gemm16<<<gg, 256, 0, stream>>>(Ap, Wp + (size_t)2 * WSLOT, bv, Vb);
  gemm16<<<gg, 256, 0, stream>>>(Ap, Wp + (size_t)3 * WSLOT, bsk, hB);  // skip into hB
  attn_logits<<<Ne / 8, 256, 0, stream>>>(Qb, Kb, ei, lg);
  fill_u32<<<(Nn + 255) / 256, 256, 0, stream>>>(mE, 0x007FFFFFu, Nn);  // enc(-inf)
  fill_f32<<<(Nn + 255) / 256, 256, 0, stream>>>(sS, 0.f, Nn);
  seg_max<<<eGrid, 256, 0, stream>>>(lg, ei, mE);
  seg_expsum<<<eGrid, 256, 0, stream>>>(lg, ei, mE, sS);
  alpha_div<<<eGrid, 256, 0, stream>>>(lg, ei, sS);
  attn_agg<<<ehGrid, 256, 0, stream>>>(lg, Vb, ei, hB);

  // ---- GatedGCN layers ----
  float* cur = hB; float* nxt = hA;
  for (int l = 0; l < Lg; ++l) {
    pack_a<<<apGrid, 256, 0, stream>>>(cur, Ap);
    gemm16<<<gg, 256, 0, stream>>>(Ap, Wp + (size_t)(4 + l)  * WSLOT, gbA + l * Hd, Qb);  // An
    gemm16<<<gg, 256, 0, stream>>>(Ap, Wp + (size_t)(7 + l)  * WSLOT, gbB + l * Hd, Kb);  // Bn
    gemm16<<<gg, 256, 0, stream>>>(Ap, Wp + (size_t)(10 + l) * WSLOT, gbC + l * Hd, Vb);  // Cn
    gemm16<<<gg, 256, 0, stream>>>(Ap, Wp + (size_t)(13 + l) * WSLOT, gbr + l * Hd, nxt); // Rn
    gated_edge<<<ehGrid, 256, 0, stream>>>(Qb, Kb, Vb, eattr, uB + l * Hd, cB + l * Hd, ei, nxt);
    relu_k<<<nhGrid, 256, 0, stream>>>(nxt, Nn * Hd);
    float* tmp = cur; cur = nxt; nxt = tmp;
  }

  // ---- fused edge scoring MLP ----
  edge_mlp<<<ETILES, 256, 0, stream>>>(cur, ei, Wp + (size_t)16 * WSLOT, mb1, mW2, mb2,
                                       (float*)d_out);
}
